// MambaTTSDecoderLayer_91311004713132
// MI455X (gfx1250) — compile-verified, matrix operand order checked by reading, not run on hardware
//
#include <hip/hip_runtime.h>

// ---------------- problem constants (match reference) ----------------
#define D_MODELc 768
#define N_HEADSc 12
#define HDc      64
#define D_FFc    3072
#define D_INNERc 1536
#define D_CONVc  4
#define DT_RANKc 48
#define DT_PADc  64
#define XPROJc   80        // DT_RANK + 2*D_STATE
#define Bc       2
#define Tc       1024
#define TFc      512
#define M1c      (Bc*Tc)   // 2048 tokens
#define MFc      (Bc*TFc)  // 1024 frame tokens

typedef __attribute__((ext_vector_type(16))) __bf16 v16bf;
typedef __attribute__((ext_vector_type(8)))  __bf16 v8bf;
typedef __attribute__((ext_vector_type(8)))  float  v8f;

enum { ACT_NONE = 0, ACT_SOFTPLUS = 1, ACT_GELU = 2 };

// ---------------- WMMA helpers ----------------
// 16-bit A-fragment layout (16x32, row = lane%16):
//   lanes 0-15 : elems 0..7 = K 0..7,   elems 8..15 = K 16..23
//   lanes 16-31: elems 0..7 = K 8..15,  elems 8..15 = K 24..31
// -> two contiguous 16B loads at (k0 + kh*8) and (k0 + 16 + kh*8).
// B fragment uses the symmetric layout with rows of W (= columns of B).
__device__ inline v16bf frag_load(const __bf16* p) {
  union { v16bf v; v8bf h[2]; } u;
  u.h[0] = *(const v8bf*)(p);
  u.h[1] = *(const v8bf*)(p + 16);
  return u.v;
}

__device__ inline v8f wmma_bf16(v16bf a, v16bf b, v8f c) {
  return __builtin_amdgcn_wmma_f32_16x16x32_bf16(false, a, false, b, (short)0, c,
                                                 false, false);
}

__device__ inline v8f vzero8() {
  v8f z = {0.f, 0.f, 0.f, 0.f, 0.f, 0.f, 0.f, 0.f};
  return z;
}

// per-lane 16B async copy global -> LDS (ASYNCcnt-tracked)
__device__ inline void async_copy_b128(unsigned lds_off, const __bf16* gptr) {
  asm volatile("global_load_async_to_lds_b128 %0, %1, off"
               :: "v"(lds_off), "v"(gptr) : "memory");
}
__device__ inline void wait_asynccnt0() {
  asm volatile("s_wait_asynccnt 0" ::: "memory");
}

__device__ inline float apply_act(float v, int act) {
  if (act == ACT_SOFTPLUS) return (v > 20.f) ? v : log1pf(__expf(v));
  if (act == ACT_GELU)     return 0.5f * v * (1.f + erff(v * 0.70710678f));
  return v;
}

// ============ LDS-staged, register-blocked GEMM (big regular shapes) =========
// C = act(A @ W^T + bias) (+resid). A: MxK bf16, W: NxK bf16.
// Requires M%64==0, N%128==0, K%64==0.
// Block = 256 threads = 8 waves. Block tile 64M x 128N; wave tile 16M x 64N.
// K staged in 64-wide chunks through double-buffered LDS (async global->LDS):
// copies for chunk i+1 overlap WMMAs on chunk i; one barrier per chunk.
#define KCH    64
#define APITCH 72    // 64 + 8 pad (144B row pitch: 16B-aligned, spreads banks)
__global__ void gemm_bf16_lds(const __bf16* __restrict__ A, int lda,
                              const __bf16* __restrict__ W, int ldw,
                              const float* __restrict__ bias,
                              const float* __restrict__ resid, int ldr,
                              float* __restrict__ C, int ldc,
                              __bf16* __restrict__ Cb,
                              int K, int act) {
  __shared__ __bf16 sA[2][64 * APITCH];    // 2 x  9.2KB
  __shared__ __bf16 sW[2][128 * APITCH];   // 2 x 18.4KB  (total ~55KB)
  const int tid  = threadIdx.x;
  const int lane = tid & 31, wave = tid >> 5;
  const int mw = wave & 3;       // M-subtile of this wave (0..3)
  const int nw = wave >> 2;      // N-half of this wave (0..1)
  const int r  = lane & 15;
  const int kh = lane >> 4;
  const int rowBase = blockIdx.y * 64;
  const int colBase = blockIdx.x * 128;

  const unsigned ldsA = (unsigned)(size_t)sA;   // flat LDS addr: low 32 bits
  const unsigned ldsW = (unsigned)(size_t)sW;
  const unsigned bufAbytes = 64 * APITCH * 2;
  const unsigned bufWbytes = 128 * APITCH * 2;

  // stage one 64-wide K chunk into buffer `buf`
  auto issue_chunk = [&](int buf, int kc) {
    // A: 64 rows x 64 elems = 512 x 16B -> 2 copies/thread
#pragma unroll
    for (int j = 0; j < 2; ++j) {
      const int idx = tid + j * 256;
      const int rowA = idx >> 3, ch = idx & 7;
      async_copy_b128(ldsA + buf * bufAbytes + (unsigned)(rowA * APITCH + ch * 8) * 2,
                      A + (size_t)(rowBase + rowA) * lda + kc + ch * 8);
    }
    // W: 128 rows x 64 elems = 1024 x 16B -> 4 copies/thread
#pragma unroll
    for (int j = 0; j < 4; ++j) {
      const int idx = tid + j * 256;
      const int rowW = idx >> 3, ch = idx & 7;
      async_copy_b128(ldsW + buf * bufWbytes + (unsigned)(rowW * APITCH + ch * 8) * 2,
                      W + (size_t)(colBase + rowW) * ldw + kc + ch * 8);
    }
  };

  v8f acc[4];
#pragma unroll
  for (int j = 0; j < 4; ++j) acc[j] = vzero8();

  issue_chunk(0, 0);
  int cur = 0;
  for (int kc = 0; kc < K; kc += KCH) {
    wait_asynccnt0();       // this wave's copies into buf `cur` landed
    __syncthreads();        // all waves' copies landed; prev compute on cur^1 done
    if (kc + KCH < K) issue_chunk(cur ^ 1, kc + KCH);   // prefetch next chunk
#pragma unroll
    for (int k0 = 0; k0 < KCH; k0 += 32) {
      const v16bf a = frag_load(&sA[cur][(mw * 16 + r) * APITCH + k0 + kh * 8]);
      v16bf bf[4];
#pragma unroll
      for (int j = 0; j < 4; ++j)
        bf[j] = frag_load(&sW[cur][(nw * 64 + j * 16 + r) * APITCH + k0 + kh * 8]);
#pragma unroll
      for (int j = 0; j < 4; ++j)
        acc[j] = wmma_bf16(a, bf[j], acc[j]);
    }
    cur ^= 1;
  }

#pragma unroll
  for (int j = 0; j < 4; ++j) {
    const int col = colBase + nw * 64 + j * 16 + r;
    const float bv = bias ? bias[col] : 0.f;
#pragma unroll
    for (int i = 0; i < 8; ++i) {
      const int row = rowBase + mw * 16 + kh * 8 + i;
      float v = apply_act(acc[j][i] + bv, act);
      if (resid) v += resid[(size_t)row * ldr + col];
      if (C)  C[(size_t)row * ldc + col]  = v;
      if (Cb) Cb[(size_t)row * ldc + col] = (__bf16)v;
    }
  }
}

// ============ generic direct-global GEMM (odd shapes: N=80, K=64) ============
__global__ void gemm_bf16(const __bf16* __restrict__ A, int lda,
                          const __bf16* __restrict__ W, int ldw,
                          const float* __restrict__ bias,
                          const float* __restrict__ resid, int ldr,
                          float* __restrict__ C, int ldc,
                          __bf16* __restrict__ Cb,
                          int M, int N, int K, int act) {
  const int lane = threadIdx.x & 31;
  const int wave = threadIdx.x >> 5;
  const int tn = blockIdx.x * 8 + wave;
  if (tn * 16 >= N) return;                       // wave-uniform early-out
  const int tm = blockIdx.y;
  const int r  = lane & 15;
  const int kh = lane >> 4;

  const __bf16* pa = A + (size_t)(tm * 16 + r) * lda + kh * 8;
  const __bf16* pw = W + (size_t)(tn * 16 + r) * ldw + kh * 8;

  v8f acc = vzero8();
  for (int k0 = 0; k0 < K; k0 += 32)
    acc = wmma_bf16(frag_load(pa + k0), frag_load(pw + k0), acc);

  const int col = tn * 16 + r;
  const float bv = bias ? bias[col] : 0.f;
#pragma unroll
  for (int i = 0; i < 8; ++i) {
    const int row = tm * 16 + kh * 8 + i;
    float v = apply_act(acc[i] + bv, act);
    if (resid) v += resid[(size_t)row * ldr + col];
    if (C)  C[(size_t)row * ldc + col]  = v;
    if (Cb) Cb[(size_t)row * ldc + col] = (__bf16)v;
  }
}

// ---------------- attention: scores = (q @ k^T) * scale ----------------
// wave computes 16 x 64 of scores (4 TF-subtiles); 8 waves cover TF=512.
__global__ void attn_scores(const __bf16* __restrict__ q,
                            const __bf16* __restrict__ k,
                            float* __restrict__ scores, float scale) {
  const int lane = threadIdx.x & 31;
  const int wave = threadIdx.x >> 5;
  const int tm = blockIdx.x;                      // T tile (0..63)
  const int z  = blockIdx.y;                      // b*H + h
  const int b = z / N_HEADSc, h = z % N_HEADSc;
  const int r = lane & 15, kh = lane >> 4;

  const __bf16* pa = q + (size_t)(b * Tc + tm * 16 + r) * D_MODELc + h * HDc + kh * 8;
  const v16bf a0 = frag_load(pa);
  const v16bf a1 = frag_load(pa + 32);

  v8f acc[4];
#pragma unroll
  for (int j = 0; j < 4; ++j) {
    const int tn = wave * 4 + j;
    const __bf16* pw = k + (size_t)(b * TFc + tn * 16 + r) * D_MODELc + h * HDc + kh * 8;
    acc[j] = wmma_bf16(a0, frag_load(pw), vzero8());
    acc[j] = wmma_bf16(a1, frag_load(pw + 32), acc[j]);
  }

  float* o = scores + (size_t)z * Tc * TFc;
#pragma unroll
  for (int j = 0; j < 4; ++j)
#pragma unroll
    for (int i = 0; i < 8; ++i)
      o[(size_t)(tm * 16 + kh * 8 + i) * TFc + (wave * 4 + j) * 16 + r] = acc[j][i] * scale;
}

// ---------------- attention: ao = attn @ V (V pre-transposed) ----------------
__global__ void attn_av(const __bf16* __restrict__ attn,
                        const __bf16* __restrict__ vt,
                        float* __restrict__ ao) {
  const int lane = threadIdx.x & 31;
  const int wave = threadIdx.x >> 5;
  const int tm = blockIdx.x * 8 + wave;           // T tile
  const int z  = blockIdx.y;
  const int b = z / N_HEADSc, h = z % N_HEADSc;
  const int r = lane & 15, kh = lane >> 4;

  const __bf16* pa = attn + (size_t)z * Tc * TFc + (size_t)(tm * 16 + r) * TFc + kh * 8;
  const __bf16* pw = vt + (size_t)z * HDc * TFc + (size_t)r * TFc + kh * 8;

  v8f acc[4];
#pragma unroll
  for (int j = 0; j < 4; ++j) acc[j] = vzero8();
  for (int k0 = 0; k0 < TFc; k0 += 32) {
    const v16bf a = frag_load(pa + k0);
    v16bf bf[4];
#pragma unroll
    for (int j = 0; j < 4; ++j)
      bf[j] = frag_load(pw + (size_t)j * 16 * TFc + k0);
#pragma unroll
    for (int j = 0; j < 4; ++j)
      acc[j] = wmma_bf16(a, bf[j], acc[j]);
  }

#pragma unroll
  for (int j = 0; j < 4; ++j)
#pragma unroll
    for (int i = 0; i < 8; ++i)
      ao[(size_t)(b * Tc + tm * 16 + kh * 8 + i) * D_MODELc + h * HDc + j * 16 + r] = acc[j][i];
}

// ---------------- layernorm (cols = 768) -> bf16 ----------------
__global__ void layernorm_bf16(const float* __restrict__ x,
                               const float* __restrict__ g,
                               const float* __restrict__ bb,
                               __bf16* __restrict__ out) {
  const int row = blockIdx.x;
  const int tid = threadIdx.x;
  const float* p = x + (size_t)row * D_MODELc;
  float v0 = p[tid], v1 = p[tid + 256], v2 = p[tid + 512];
  __shared__ float red[256];
  red[tid] = v0 + v1 + v2;
  __syncthreads();
  for (int o = 128; o > 0; o >>= 1) { if (tid < o) red[tid] += red[tid + o]; __syncthreads(); }
  const float mu = red[0] * (1.f / D_MODELc);
  __syncthreads();
  const float d0 = v0 - mu, d1 = v1 - mu, d2 = v2 - mu;
  red[tid] = d0 * d0 + d1 * d1 + d2 * d2;
  __syncthreads();
  for (int o = 128; o > 0; o >>= 1) { if (tid < o) red[tid] += red[tid + o]; __syncthreads(); }
  const float rs = rsqrtf(red[0] * (1.f / D_MODELc) + 1e-5f);
  __bf16* o = out + (size_t)row * D_MODELc;
  o[tid]       = (__bf16)(d0 * rs * g[tid]       + bb[tid]);
  o[tid + 256] = (__bf16)(d1 * rs * g[tid + 256] + bb[tid + 256]);
  o[tid + 512] = (__bf16)(d2 * rs * g[tid + 512] + bb[tid + 512]);
}

// ---------------- causal depthwise conv (width 4) + SiLU ----------------
__global__ void conv_silu(const float* __restrict__ xz,   // (B,T,2*D_INNER)
                          const float* __restrict__ cw,   // (1536,4)
                          const float* __restrict__ cb,
                          float* __restrict__ u, __bf16* __restrict__ ub) {
  const size_t i = (size_t)blockIdx.x * 256 + threadIdx.x;   // over B*T*D_INNER
  const int c = (int)(i % D_INNERc);
  const size_t bt = i / D_INNERc;
  const int t = (int)(bt % Tc);
  const size_t b = bt / Tc;
  float acc = cb[c];
#pragma unroll
  for (int j = 0; j < D_CONVc; ++j) {
    const int tt = t - (D_CONVc - 1) + j;
    if (tt >= 0)
      acc += xz[((size_t)(b * Tc + tt)) * (2 * D_INNERc) + c] * cw[c * D_CONVc + j];
  }
  const float s = acc / (1.f + __expf(-acc));               // SiLU
  u[i] = s;
  ub[i] = (__bf16)s;
}

// ---------------- selective scan ----------------
__global__ void mamba_scan(const float* __restrict__ dtsp,   // (B,T,1536)
                           const float* __restrict__ u,      // (B,T,1536)
                           const float* __restrict__ xdb,    // (B,T,80)
                           const float* __restrict__ A_log,  // (1536,16)
                           const float* __restrict__ Dv,     // (1536)
                           float* __restrict__ y) {
  const int b = blockIdx.y;
  const int n = blockIdx.x * 256 + threadIdx.x;
  __shared__ float sB[16], sC[16];
  float Ar[16];
#pragma unroll
  for (int s = 0; s < 16; ++s) Ar[s] = -__expf(A_log[n * 16 + s]);
  float hst[16];
#pragma unroll
  for (int s = 0; s < 16; ++s) hst[s] = 0.f;
  const float Dn = Dv[n];
  for (int t = 0; t < Tc; ++t) {
    const size_t bt = (size_t)b * Tc + t;
    if (threadIdx.x < 32) {
      const int s = threadIdx.x & 15;
      const float v = xdb[bt * XPROJc + DT_RANKc + (threadIdx.x < 16 ? 0 : 16) + s];
      if (threadIdx.x < 16) sB[s] = v; else sC[s] = v;
    }
    __syncthreads();
    const float dt = dtsp[bt * D_INNERc + n];
    const float uu = u[bt * D_INNERc + n];
    float acc = 0.f;
#pragma unroll
    for (int s = 0; s < 16; ++s) {
      const float h2 = __expf(dt * Ar[s]) * hst[s] + dt * sB[s] * uu;
      hst[s] = h2;
      acc += h2 * sC[s];
    }
    y[bt * D_INNERc + n] = acc + uu * Dn;
    __syncthreads();
  }
}

// ---------------- y = y * silu(z) -> bf16 ----------------
__global__ void gate_silu(const float* __restrict__ y, const float* __restrict__ xz,
                          __bf16* __restrict__ yb) {
  const size_t i = (size_t)blockIdx.x * 256 + threadIdx.x;
  const int c = (int)(i % D_INNERc);
  const size_t bt = i / D_INNERc;
  const float z = xz[bt * (2 * D_INNERc) + D_INNERc + c];
  yb[i] = (__bf16)(y[i] * (z / (1.f + __expf(-z))));
}

// ---------------- masked softmax over TF=512, output bf16 ----------------
__global__ void softmax_rows(const float* __restrict__ scores,
                             const unsigned char* __restrict__ mask,
                             __bf16* __restrict__ attn) {
  const int row = blockIdx.x;                 // (b*H + h)*T + t
  const int tid = threadIdx.x;
  const int b = row / (N_HEADSc * Tc);
  const float* s = scores + (size_t)row * TFc;
  const unsigned char* m = mask + (size_t)b * TFc;
  __shared__ float red[256];
  const int j0 = tid, j1 = tid + 256;
  const bool m0 = m[j0] != 0, m1 = m[j1] != 0;
  const float NEG = -3.402823e38f;
  const float s0 = m0 ? s[j0] : NEG;
  const float s1 = m1 ? s[j1] : NEG;
  red[tid] = fmaxf(s0, s1);
  __syncthreads();
  for (int o = 128; o > 0; o >>= 1) { if (tid < o) red[tid] = fmaxf(red[tid], red[tid + o]); __syncthreads(); }
  const float mx = red[0];
  __syncthreads();
  const float e0 = m0 ? __expf(s0 - mx) : 0.f;
  const float e1 = m1 ? __expf(s1 - mx) : 0.f;
  red[tid] = e0 + e1;
  __syncthreads();
  for (int o = 128; o > 0; o >>= 1) { if (tid < o) red[tid] += red[tid + o]; __syncthreads(); }
  const float inv = 1.f / (red[0] + 1e-20f);
  __bf16* a = attn + (size_t)row * TFc;
  a[j0] = (__bf16)(e0 * inv);
  a[j1] = (__bf16)(e1 * inv);
}

// ---------------- small converters ----------------
__global__ void cvt_bf16(const float* __restrict__ s, __bf16* __restrict__ d, long n) {
  const long i = (long)blockIdx.x * 256 + threadIdx.x;
  if (i < n) d[i] = (__bf16)s[i];
}

__global__ void pack_pad_bf16(const float* __restrict__ src, int sld,
                              __bf16* __restrict__ dst, int dld, int cols, long rows) {
  const long i = (long)blockIdx.x * 256 + threadIdx.x;
  if (i >= rows * (long)dld) return;
  const int c = (int)(i % dld);
  const long r = i / dld;
  dst[i] = (c < cols) ? (__bf16)src[r * (long)sld + c] : (__bf16)0.f;
}

// v (B,TF,H*64) f32 -> vt (B*H, 64, TF) bf16
__global__ void transpose_v(const float* __restrict__ v, __bf16* __restrict__ vt) {
  const long i = (long)blockIdx.x * 256 + threadIdx.x;
  const int t = (int)(i % TFc);
  long r = i / TFc;
  const int d = (int)(r % HDc);
  r /= HDc;
  const int h = (int)(r % N_HEADSc);
  const int b = (int)(r / N_HEADSc);
  vt[i] = (__bf16)v[((size_t)b * TFc + t) * D_MODELc + h * HDc + d];
}

// ---------------- host orchestration ----------------
static inline dim3 g1d(long n) { return dim3((unsigned)((n + 255) / 256)); }

extern "C" void kernel_launch(void* const* d_in, const int* in_sizes, int n_in,
                              void* d_out, int out_size, void* d_ws, size_t ws_size,
                              hipStream_t stream) {
  (void)in_sizes; (void)n_in; (void)out_size; (void)ws_size;
  const float* x        = (const float*)d_in[0];
  const float* styled   = (const float*)d_in[1];
  const unsigned char* smask = (const unsigned char*)d_in[2];
  const float* ln1_g = (const float*)d_in[3];
  const float* ln1_b = (const float*)d_in[4];
  const float* ln2_g = (const float*)d_in[5];
  const float* ln2_b = (const float*)d_in[6];
  const float* ln3_g = (const float*)d_in[7];
  const float* ln3_b = (const float*)d_in[8];
  const float* in_proj_w  = (const float*)d_in[9];
  const float* conv_w     = (const float*)d_in[10];
  const float* conv_b     = (const float*)d_in[11];
  const float* x_proj_w   = (const float*)d_in[12];
  const float* dt_proj_w  = (const float*)d_in[13];
  const float* dt_proj_b  = (const float*)d_in[14];
  const float* A_log      = (const float*)d_in[15];
  const float* Dvec       = (const float*)d_in[16];
  const float* out_proj_w = (const float*)d_in[17];
  const float* attn_in_w  = (const float*)d_in[18];
  const float* attn_in_b  = (const float*)d_in[19];
  const float* attn_out_w = (const float*)d_in[20];
  const float* attn_out_b = (const float*)d_in[21];
  const float* ff_w1 = (const float*)d_in[22];
  const float* ff_b1 = (const float*)d_in[23];
  const float* ff_w2 = (const float*)d_in[24];
  const float* ff_b2 = (const float*)d_in[25];
  float* out = (float*)d_out;

  // ---- workspace layout: persistent region + phase-aliased scratch region ----
  char* base = (char*)d_ws;
  size_t off = 0;
  auto P = [&](size_t bytes) -> char* {
    size_t o = (off + 255) & ~(size_t)255; off = o + bytes; return base + o;
  };
  __bf16* inWb   = (__bf16*)P((size_t)(2 * D_INNERc) * D_MODELc * 2);
  __bf16* xpWb   = (__bf16*)P((size_t)XPROJc * D_INNERc * 2);
  __bf16* dtWb   = (__bf16*)P((size_t)D_INNERc * DT_PADc * 2);
  __bf16* outWb  = (__bf16*)P((size_t)D_MODELc * D_INNERc * 2);
  __bf16* attnWb = (__bf16*)P((size_t)(3 * D_MODELc) * D_MODELc * 2);
  __bf16* aoWb   = (__bf16*)P((size_t)D_MODELc * D_MODELc * 2);
  __bf16* f1Wb   = (__bf16*)P((size_t)D_FFc * D_MODELc * 2);
  __bf16* f2Wb   = (__bf16*)P((size_t)D_MODELc * D_FFc * 2);
  __bf16* sfbf   = (__bf16*)P((size_t)MFc * D_MODELc * 2);
  __bf16* hbf    = (__bf16*)P((size_t)M1c * D_MODELc * 2);
  char* S = base + ((off + 255) & ~(size_t)255);

  // phase 1 (mamba) scratch
  size_t o1 = 0;
  auto G1 = [&](size_t bytes) -> char* {
    size_t o = (o1 + 255) & ~(size_t)255; o1 = o + bytes; return S + o;
  };
  float*  xz    = (float*)G1((size_t)M1c * 2 * D_INNERc * 4);
  float*  uconv = (float*)G1((size_t)M1c * D_INNERc * 4);
  __bf16* ubf   = (__bf16*)G1((size_t)M1c * D_INNERc * 2);
  float*  xdb   = (float*)G1((size_t)M1c * XPROJc * 4);
  __bf16* dtpad = (__bf16*)G1((size_t)M1c * DT_PADc * 2);
  float*  dtsp  = (float*)G1((size_t)M1c * D_INNERc * 4);
  float*  yb    = (float*)G1((size_t)M1c * D_INNERc * 4);
  __bf16* ybf   = (__bf16*)G1((size_t)M1c * D_INNERc * 2);

  // phase 2 (attention) scratch, aliases phase 1
  size_t o2 = 0;
  auto G2 = [&](size_t bytes) -> char* {
    size_t o = (o2 + 255) & ~(size_t)255; o2 = o + bytes; return S + o;
  };
  __bf16* qbf    = (__bf16*)G2((size_t)M1c * D_MODELc * 2);
  __bf16* kbf    = (__bf16*)G2((size_t)MFc * D_MODELc * 2);
  float*  vf     = (float*)G2((size_t)MFc * D_MODELc * 4);
  __bf16* vtbf   = (__bf16*)G2((size_t)Bc * N_HEADSc * HDc * TFc * 2);
  float*  scores = (float*)G2((size_t)Bc * N_HEADSc * Tc * TFc * 4);
  __bf16* attnbf = (__bf16*)G2((size_t)Bc * N_HEADSc * Tc * TFc * 2);
  float*  ao     = (float*)G2((size_t)M1c * D_MODELc * 4);
  __bf16* aobf   = (__bf16*)G2((size_t)M1c * D_MODELc * 2);

  // phase 3 (FFN) scratch, aliases again
  __bf16* ffhbf = (__bf16*)S;

  const dim3 B256(256);

  // ---- convert weights to bf16 (every launch; no cached state) ----
  cvt_bf16<<<g1d((long)2 * D_INNERc * D_MODELc), B256, 0, stream>>>(in_proj_w, inWb, (long)2 * D_INNERc * D_MODELc);
  cvt_bf16<<<g1d((long)XPROJc * D_INNERc), B256, 0, stream>>>(x_proj_w, xpWb, (long)XPROJc * D_INNERc);
  pack_pad_bf16<<<g1d((long)D_INNERc * DT_PADc), B256, 0, stream>>>(dt_proj_w, DT_RANKc, dtWb, DT_PADc, DT_RANKc, D_INNERc);
  cvt_bf16<<<g1d((long)D_MODELc * D_INNERc), B256, 0, stream>>>(out_proj_w, outWb, (long)D_MODELc * D_INNERc);
  cvt_bf16<<<g1d((long)3 * D_MODELc * D_MODELc), B256, 0, stream>>>(attn_in_w, attnWb, (long)3 * D_MODELc * D_MODELc);
  cvt_bf16<<<g1d((long)D_MODELc * D_MODELc), B256, 0, stream>>>(attn_out_w, aoWb, (long)D_MODELc * D_MODELc);
  cvt_bf16<<<g1d((long)D_FFc * D_MODELc), B256, 0, stream>>>(ff_w1, f1Wb, (long)D_FFc * D_MODELc);
  cvt_bf16<<<g1d((long)D_MODELc * D_FFc), B256, 0, stream>>>(ff_w2, f2Wb, (long)D_MODELc * D_FFc);
  cvt_bf16<<<g1d((long)MFc * D_MODELc), B256, 0, stream>>>(styled, sfbf, (long)MFc * D_MODELc);

  // residual stream lives in d_out
  hipMemcpyAsync(out, x, (size_t)M1c * D_MODELc * sizeof(float), hipMemcpyDeviceToDevice, stream);

  // ================= Mamba block =================
  layernorm_bf16<<<dim3(M1c), B256, 0, stream>>>(x, ln1_g, ln1_b, hbf);
  // xz = h @ in_proj_w^T   (2048 x 3072, K=768)
  gemm_bf16_lds<<<dim3(3072 / 128, M1c / 64), B256, 0, stream>>>(hbf, D_MODELc, inWb, D_MODELc,
      nullptr, nullptr, 0, xz, 2 * D_INNERc, nullptr, D_MODELc, ACT_NONE);
  conv_silu<<<g1d((long)M1c * D_INNERc), B256, 0, stream>>>(xz, conv_w, conv_b, uconv, ubf);
  // xdb = u @ x_proj_w^T   (2048 x 80, K=1536) - odd N, direct kernel
  gemm_bf16<<<dim3(1, M1c / 16), B256, 0, stream>>>(ubf, D_INNERc, xpWb, D_INNERc,
      nullptr, nullptr, 0, xdb, XPROJc, nullptr, M1c, XPROJc, D_INNERc, ACT_NONE);
  pack_pad_bf16<<<g1d((long)M1c * DT_PADc), B256, 0, stream>>>(xdb, XPROJc, dtpad, DT_PADc, DT_RANKc, M1c);
  // dt = softplus(dt @ dt_proj_w^T + b)   (2048 x 1536, K=64) - odd K, direct kernel
  gemm_bf16<<<dim3(12, M1c / 16), B256, 0, stream>>>(dtpad, DT_PADc, dtWb, DT_PADc,
      dt_proj_b, nullptr, 0, dtsp, D_INNERc, nullptr, M1c, D_INNERc, DT_PADc, ACT_SOFTPLUS);
  mamba_scan<<<dim3(D_INNERc / 256, Bc), B256, 0, stream>>>(dtsp, uconv, xdb, A_log, Dvec, yb);
  gate_silu<<<g1d((long)M1c * D_INNERc), B256, 0, stream>>>(yb, xz, ybf);
  // x += y @ out_proj_w^T
  gemm_bf16_lds<<<dim3(D_MODELc / 128, M1c / 64), B256, 0, stream>>>(ybf, D_INNERc, outWb, D_INNERc,
      nullptr, out, D_MODELc, out, D_MODELc, nullptr, D_INNERc, ACT_NONE);

  // ================= Cross-attention =================
  layernorm_bf16<<<dim3(M1c), B256, 0, stream>>>(out, ln2_g, ln2_b, hbf);
  // q = h @ Wq^T + bq (bf16 only)
  gemm_bf16_lds<<<dim3(D_MODELc / 128, M1c / 64), B256, 0, stream>>>(hbf, D_MODELc, attnWb, D_MODELc,
      attn_in_b, nullptr, 0, nullptr, D_MODELc, qbf, D_MODELc, ACT_NONE);
  // k = styled @ Wk^T + bk
  gemm_bf16_lds<<<dim3(D_MODELc / 128, MFc / 64), B256, 0, stream>>>(sfbf, D_MODELc,
      attnWb + (size_t)D_MODELc * D_MODELc, D_MODELc,
      attn_in_b + D_MODELc, nullptr, 0, nullptr, D_MODELc, kbf, D_MODELc, ACT_NONE);
  // v = styled @ Wv^T + bv (f32, for transpose)
  gemm_bf16_lds<<<dim3(D_MODELc / 128, MFc / 64), B256, 0, stream>>>(sfbf, D_MODELc,
      attnWb + (size_t)2 * D_MODELc * D_MODELc, D_MODELc,
      attn_in_b + 2 * D_MODELc, nullptr, 0, vf, D_MODELc, nullptr, D_MODELc, ACT_NONE);
  transpose_v<<<g1d((long)Bc * N_HEADSc * HDc * TFc), B256, 0, stream>>>(vf, vtbf);
  attn_scores<<<dim3(Tc / 16, Bc * N_HEADSc), B256, 0, stream>>>(qbf, kbf, scores, 0.125f);
  softmax_rows<<<dim3(Bc * N_HEADSc * Tc), B256, 0, stream>>>(scores, smask, attnbf);
  attn_av<<<dim3(Tc / 16 / 8, Bc * N_HEADSc), B256, 0, stream>>>(attnbf, vtbf, ao);
  cvt_bf16<<<g1d((long)M1c * D_MODELc), B256, 0, stream>>>(ao, aobf, (long)M1c * D_MODELc);
  // x += ao @ attn_out_w^T + b
  gemm_bf16_lds<<<dim3(D_MODELc / 128, M1c / 64), B256, 0, stream>>>(aobf, D_MODELc, aoWb, D_MODELc,
      attn_out_b, out, D_MODELc, out, D_MODELc, nullptr, D_MODELc, ACT_NONE);

  // ================= FFN =================
  layernorm_bf16<<<dim3(M1c), B256, 0, stream>>>(out, ln3_g, ln3_b, hbf);
  // h1 = gelu(h @ W1^T + b1) -> bf16
  gemm_bf16_lds<<<dim3(D_FFc / 128, M1c / 64), B256, 0, stream>>>(hbf, D_MODELc, f1Wb, D_MODELc,
      ff_b1, nullptr, 0, nullptr, D_FFc, ffhbf, D_MODELc, ACT_GELU);
  // x += h1 @ W2^T + b2
  gemm_bf16_lds<<<dim3(D_MODELc / 128, M1c / 64), B256, 0, stream>>>(ffhbf, D_FFc, f2Wb, D_FFc,
      ff_b2, out, D_MODELc, out, D_MODELc, nullptr, D_FFc, ACT_NONE);
}
